// RNNModel_22247930593895
// MI455X (gfx1250) — matrix operand order repaired and checked by reading, not compiled, not run on hardware
//
#include <hip/hip_runtime.h>
#include <hip/hip_bf16.h>
#include <stdint.h>

// ---------------- problem constants (from reference) ----------------
#define S_LEN 128
#define BATCH 32
#define VOCAB 32000
#define DIM   1024
#define HID   1024
#define G4H   4096           // 4*HID
#define NBLK_SCAN 64         // persistent-scan blocks (512 waves total)
#define LDSLD 1032           // padded LDS row (1024 + 8 ushorts) to spread banks

// ---------------- types ----------------
typedef __attribute__((ext_vector_type(16))) __bf16 bf16x16;
typedef __attribute__((ext_vector_type(8)))  float  floatx8;
typedef __attribute__((ext_vector_type(8)))  unsigned int uintx8;

// round-to-nearest-even f32 -> bf16 (as raw u16)
__device__ __forceinline__ unsigned short f2bf(float f) {
  unsigned int u = __builtin_bit_cast(unsigned int, f);
  u += 0x7FFFu + ((u >> 16) & 1u);
  return (unsigned short)(u >> 16);
}

__device__ __forceinline__ float sigm(float x) {
  return 1.f / (1.f + __expf(-x));
}

// ---------------- fragment loaders (wave32 WMMA bf16 layouts, ISA 7.12.2) ----------------
// A: 16x32 (MxK); lane row m=lane&15; lanes0-15: K {0..7,16..23}, lanes16-31: K {8..15,24..31}
__device__ __forceinline__ bf16x16 load_frag_a(const unsigned short* __restrict__ base,
                                               int ld, int mbase, int kbase, int lane) {
  int m = lane & 15, hi = lane >> 4;
  const unsigned short* p = base + (size_t)(mbase + m) * ld + kbase + hi * 8;
  uint4 c0 = *(const uint4*)p;
  uint4 c1 = *(const uint4*)(p + 16);
  uintx8 r = { c0.x, c0.y, c0.z, c0.w, c1.x, c1.y, c1.z, c1.w };
  return __builtin_bit_cast(bf16x16, r);
}

// B: 32x16 (KxN) with B[k][n] = W[n][k] (W row-major [N,K]); lane col n=lane&15,
// lanes0-15: K {0..7,16..23}, lanes16-31: K {8..15,24..31}
__device__ __forceinline__ bf16x16 load_frag_b(const unsigned short* w,
                                               int ld, int nbase, int kbase, int lane) {
  int n = lane & 15, hi = lane >> 4;
  const unsigned short* p = w + (size_t)(nbase + n) * ld + kbase + hi * 8;
  uint4 c0 = *(const uint4*)p;
  uint4 c1 = *(const uint4*)(p + 16);
  uintx8 r = { c0.x, c0.y, c0.z, c0.w, c1.x, c1.y, c1.z, c1.w };
  return __builtin_bit_cast(bf16x16, r);
}

#define WMMA_BF16(a, b, acc) \
  __builtin_amdgcn_wmma_f32_16x16x32_bf16(false, (a), false, (b), (short)0, (acc), false, false)

// ---------------- conversion kernels ----------------
__global__ void k_f32_to_bf16(const float* __restrict__ in,
                              unsigned short* __restrict__ out, int n) {
  int i = blockIdx.x * blockDim.x + threadIdx.x;
  if (i < n) out[i] = f2bf(in[i]);
}

// gate-interleaved weight convert: out row 4j+g = in row g*HID + j  (K = 1024)
__global__ void k_conv_gatew(const float* __restrict__ in,
                             unsigned short* __restrict__ out) {
  size_t i = (size_t)blockIdx.x * blockDim.x + threadIdx.x;  // over 4H*1024
  int k = (int)(i & 1023);
  int p = (int)(i >> 10);
  int g = p & 3, j = p >> 2;
  out[i] = f2bf(in[((size_t)g * HID + j) * 1024 + k]);
}

// embedding gather + bf16: out[s,b,d] = bf16(embW[input[s,b], d])
__global__ void k_embed_gather(const int* __restrict__ idx,
                               const float* __restrict__ embW,
                               unsigned short* __restrict__ out) {
  int i = blockIdx.x * blockDim.x + threadIdx.x;  // S*B*D
  int d  = i & (DIM - 1);
  int sb = i >> 10;
  int row = idx[sb];
  out[i] = f2bf(embW[(size_t)row * DIM + d]);
}

// ---------------- generic GEMM: out[M,ldout] = A[M,1024] @ W[N,1024]^T (+bias) ----------------
// grid(x = N/64, y = M/128); each wave: 16(M) x 64(N) tile, 4 accumulators, A reused 4x
__global__ void __launch_bounds__(256) k_gemm_bf16(
    const unsigned short* __restrict__ A,
    const unsigned short* __restrict__ W,
    const float* __restrict__ bias,      // may be null
    float* __restrict__ out, int ldout) {
  int wave = threadIdx.x >> 5;
  int lane = threadIdx.x & 31;
  int mbase  = (blockIdx.y * 8 + wave) * 16;
  int nbase0 = blockIdx.x * 64;

  floatx8 a0 = {}, a1 = {}, a2 = {}, a3 = {};
#pragma unroll 2
  for (int k = 0; k < HID; k += 32) {
    bf16x16 a  = load_frag_a(A, HID, mbase, k, lane);
    bf16x16 b0 = load_frag_b(W, HID, nbase0,      k, lane);
    bf16x16 b1 = load_frag_b(W, HID, nbase0 + 16, k, lane);
    bf16x16 b2 = load_frag_b(W, HID, nbase0 + 32, k, lane);
    bf16x16 b3 = load_frag_b(W, HID, nbase0 + 48, k, lane);
    a0 = WMMA_BF16(a, b0, a0);
    a1 = WMMA_BF16(a, b1, a1);
    a2 = WMMA_BF16(a, b2, a2);
    a3 = WMMA_BF16(a, b3, a3);
  }

  int n = lane & 15, hi = lane >> 4;
  float bb0 = 0.f, bb1 = 0.f, bb2 = 0.f, bb3 = 0.f;
  if (bias) {
    bb0 = bias[nbase0 + n];
    bb1 = bias[nbase0 + 16 + n];
    bb2 = bias[nbase0 + 32 + n];
    bb3 = bias[nbase0 + 48 + n];
  }
#pragma unroll
  for (int v = 0; v < 8; ++v) {
    size_t row = (size_t)(mbase + hi * 8 + v);
    float* o = out + row * ldout + nbase0 + n;
    o[0]  = a0[v] + bb0;
    o[16] = a1[v] + bb1;
    o[32] = a2[v] + bb2;
    o[48] = a3[v] + bb3;
  }
}

// ---------------- persistent LSTM scan: one launch for all 128 steps ----------------
// 64 blocks x 8 waves = 512 waves; wave tile = 16 batch x 16 gate-cols (= 4 hidden units x 4 gates).
// w_hh slices (64 rows / block = 128KB) staged once into dynamic LDS (CDNA5: 320KB/WGP).
// Per step: WMMA over K=HID from LDS-resident weights, add precomputed XW[t], apply the
// LSTM cell in-register via 4-lane gate shuffles, publish h_t (bf16), one grid barrier.
__global__ void __launch_bounds__(256) k_lstm_scan(
    const float* __restrict__ XW,            // [S, 32, 4H] f32, gate-interleaved cols
    const unsigned short* __restrict__ whh,  // [4H, HID] bf16, gate-interleaved rows
    const unsigned short* __restrict__ h0bf, // [32, HID] bf16
    unsigned short* __restrict__ hidbf,      // [S, 32, HID] bf16 (scan output)
    float* __restrict__ c,                   // [32, HID] f32 state
    float* __restrict__ hf,                  // [32, HID] f32 (final h)
    unsigned int* __restrict__ bar) {        // grid barrier counter (pre-zeroed)
  extern __shared__ unsigned short smem[];   // 64 rows x LDSLD ushorts

  int wave = threadIdx.x >> 5;
  int lane = threadIdx.x & 31;
  int tile  = blockIdx.x * 8 + wave;         // 0..511
  int mbase = (tile & 1) * 16;               // batch tile: rows 0..15 / 16..31
  int nbase = (tile >> 1) * 16;              // gate-col tile
  int n = lane & 15, hi = lane >> 4;
  int gate = n & 3, ju = n >> 2;
  int j = (nbase >> 2) + ju;                 // hidden-unit index of this lane
  int lgrp = lane & ~3;                      // 4-lane gate group base

  // ---- stage this block's 64 w_hh rows into LDS (rows blk*64 .. blk*64+63) ----
  int gRowBase = blockIdx.x * 64;
  for (int idx = threadIdx.x; idx < 64 * 128; idx += 256) {  // 128 uint4 per row
    int r = idx >> 7, q = idx & 127;
    uint4 v = *((const uint4*)(whh + (size_t)(gRowBase + r) * HID) + q);
    *(uint4*)(&smem[(size_t)r * LDSLD + q * 8]) = v;
  }
  __syncthreads();
  int ldsRowBase = (wave >> 1) * 16;         // this wave's 16 weight rows in LDS

  for (int t = 0; t < S_LEN; ++t) {
    const unsigned short* hp =
        t ? hidbf + (size_t)(t - 1) * BATCH * HID : h0bf;

    floatx8 acc = {};
#pragma unroll 4
    for (int k = 0; k < HID; k += 32) {
      bf16x16 a = load_frag_a(hp, HID, mbase, k, lane);
      bf16x16 b = load_frag_b(smem, LDSLD, ldsRowBase, k, lane);  // ds_load path
      acc = WMMA_BF16(a, b, acc);
    }

    // add precomputed input projection
    const float* xwt = XW + (size_t)t * BATCH * G4H;
#pragma unroll
    for (int v = 0; v < 8; ++v)
      acc[v] += xwt[(size_t)(mbase + hi * 8 + v) * G4H + nbase + n];

    // in-register LSTM cell: each 4-lane group holds i,f,g,o of one hidden unit
    unsigned short* hbt = hidbf + (size_t)t * BATCH * HID;
#pragma unroll
    for (int v = 0; v < 8; ++v) {
      float gi = __shfl(acc[v], lgrp + 0, 32);
      float gf = __shfl(acc[v], lgrp + 1, 32);
      float gg = __shfl(acc[v], lgrp + 2, 32);
      float go = __shfl(acc[v], lgrp + 3, 32);
      if ((v >> 1) == gate) {                // distribute 8 rows over the 4 lanes
        int row = mbase + hi * 8 + v;
        size_t ci = (size_t)row * HID + j;
        float cn = sigm(gf) * c[ci] + sigm(gi) * tanhf(gg);
        float hn = sigm(go) * tanhf(cn);
        c[ci] = cn;
        hbt[ci] = f2bf(hn);
        if (t == S_LEN - 1) hf[ci] = hn;
      }
    }

    // ---- grid barrier: publish h_t before any block starts step t+1 ----
    __threadfence();
    __syncthreads();
    if (threadIdx.x == 0) {
      __hip_atomic_fetch_add(bar, 1u, __ATOMIC_RELEASE, __HIP_MEMORY_SCOPE_AGENT);
      unsigned int target = (unsigned int)(NBLK_SCAN * (t + 1));
      while (__hip_atomic_load(bar, __ATOMIC_ACQUIRE, __HIP_MEMORY_SCOPE_AGENT) < target)
        __builtin_amdgcn_s_sleep(1);
    }
    __syncthreads();
  }
}

// ---------------- host orchestration ----------------
extern "C" void kernel_launch(void* const* d_in, const int* in_sizes, int n_in,
                              void* d_out, int out_size, void* d_ws, size_t ws_size,
                              hipStream_t stream) {
  const int*   input = (const int*)d_in[0];    // [S,B]
  const float* h0    = (const float*)d_in[1];  // [1,B,H]
  const float* c0    = (const float*)d_in[2];  // [1,B,H]
  const float* embW  = (const float*)d_in[3];  // [V,D]
  const float* w_ih  = (const float*)d_in[4];  // [4H,D]
  const float* w_hh  = (const float*)d_in[5];  // [4H,H]
  const float* dec_w = (const float*)d_in[6];  // [V,H]
  const float* dec_b = (const float*)d_in[7];  // [V]
  float* out = (float*)d_out;                  // decoded | h | c

  // workspace carve-up
  char* ws = (char*)d_ws;
  unsigned short* emb_bf = (unsigned short*)ws; ws += (size_t)S_LEN * BATCH * DIM * 2;  // 8 MB
  unsigned short* wih_bf = (unsigned short*)ws; ws += (size_t)G4H * DIM * 2;            // 8 MB
  unsigned short* whh_bf = (unsigned short*)ws; ws += (size_t)G4H * HID * 2;            // 8 MB
  unsigned short* dcw_bf = (unsigned short*)ws; ws += (size_t)VOCAB * HID * 2;          // 64 MB
  unsigned short* hid_bf = (unsigned short*)ws; ws += (size_t)S_LEN * BATCH * HID * 2;  // 8 MB
  unsigned short* h0_bf  = (unsigned short*)ws; ws += (size_t)BATCH * HID * 2;
  float* xw_ws = (float*)ws; ws += (size_t)S_LEN * BATCH * G4H * 4;                     // 64 MB
  float* c_ws  = (float*)ws; ws += (size_t)BATCH * HID * 4;
  float* hf_ws = (float*)ws; ws += (size_t)BATCH * HID * 4;
  unsigned int* bar_ws = (unsigned int*)ws; ws += 256;

  // one-time conversions (gate-interleaved weights for the fused cell)
  k_conv_gatew<<<(G4H * DIM) / 256, 256, 0, stream>>>(w_ih, wih_bf);
  k_conv_gatew<<<(G4H * HID) / 256, 256, 0, stream>>>(w_hh, whh_bf);
  k_f32_to_bf16<<<(VOCAB * HID) / 256, 256, 0, stream>>>(dec_w, dcw_bf, VOCAB * HID);
  k_f32_to_bf16<<<(BATCH * HID) / 256, 256, 0, stream>>>(h0, h0_bf, BATCH * HID);
  k_embed_gather<<<(S_LEN * BATCH * DIM) / 256, 256, 0, stream>>>(input, embW, emb_bf);
  hipMemcpyAsync(c_ws, c0, (size_t)BATCH * HID * sizeof(float),
                 hipMemcpyDeviceToDevice, stream);
  hipMemsetAsync(bar_ws, 0, sizeof(unsigned int), stream);

  // hoisted input projection for all steps: XW = emb @ w_ih^T  (parallel GEMM)
  {
    dim3 g(G4H / 64, (S_LEN * BATCH / 16) / 8);  // (64, 32)
    k_gemm_bf16<<<g, 256, 0, stream>>>(emb_bf, wih_bf, nullptr, xw_ws, G4H);
  }

  // persistent recurrent scan (single launch, LDS-resident w_hh, 1 barrier/step)
  k_lstm_scan<<<NBLK_SCAN, 256, 64 * LDSLD * 2, stream>>>(
      xw_ws, whh_bf, h0_bf, hid_bf, c_ws, hf_ws, bar_ws);

  // decoder over all timesteps: [S*B, H] @ [H, V] + b
  {
    dim3 g(VOCAB / 64, (S_LEN * BATCH / 16) / 8);  // (500, 32)
    k_gemm_bf16<<<g, 256, 0, stream>>>(hid_bf, dcw_bf, dec_b, out, VOCAB);
  }

  // final h, c appended after decoded
  size_t dec_elems = (size_t)S_LEN * BATCH * VOCAB;
  hipMemcpyAsync(out + dec_elems, hf_ws, (size_t)BATCH * HID * sizeof(float),
                 hipMemcpyDeviceToDevice, stream);
  hipMemcpyAsync(out + dec_elems + BATCH * HID, c_ws,
                 (size_t)BATCH * HID * sizeof(float),
                 hipMemcpyDeviceToDevice, stream);
}
